// MambaClassification_47519518163311
// MI455X (gfx1250) — matrix operand order repaired and checked by reading, not compiled, not run on hardware
//
#include <hip/hip_runtime.h>
#include <hip/hip_bf16.h>
#include <math.h>

// ---------------------------------------------------------------------------
// Mamba classifier forward for MI455X (gfx1250, wave32, WMMA).
// GEMMs: v_wmma_f32_16x16x32_bf16 (bf16 in, f32 accumulate), branch-free
// inner loop. Scan: async global->LDS broadcast (ASYNCcnt) per time step.
// ---------------------------------------------------------------------------

typedef __attribute__((ext_vector_type(16))) __bf16 v16bf;
typedef __attribute__((ext_vector_type(8)))  float  v8f;

#define BB   2
#define LL   2048
#define NCC  3
#define DMC  256
#define DM   768          // NCC*DMC
#define DI   1536         // 2*DM
#define DSS  16
#define DRR  48
#define DCC  4
#define NLL  4
#define EPSF 1e-5f

#define USE_ASYNC_LDS 1   // global_load_async_to_lds_b128 path in scan

// ---------------------------------------------------------------------------
// Embedding gather: x[b,l, c*256+j] = emb[tokens[b,l,c], j]
// ---------------------------------------------------------------------------
__global__ void __launch_bounds__(256) k_embed(const int* __restrict__ tokens,
                                               const float* __restrict__ emb,
                                               float* __restrict__ x) {
  int idx = blockIdx.x * 256 + threadIdx.x;            // over BB*LL*NCC*DMC
  if (idx >= BB * LL * NCC * DMC) return;
  int j = idx % DMC;
  int c = (idx / DMC) % NCC;
  int t = idx / (DMC * NCC);                           // b*LL + l
  int tok = tokens[t * NCC + c];
  x[(size_t)t * DM + c * DMC + j] = emb[(size_t)tok * DMC + j];
}

// ---------------------------------------------------------------------------
// RMSNorm over rows of length DM. One block per row.
// ---------------------------------------------------------------------------
__global__ void __launch_bounds__(256) k_rmsnorm(const float* __restrict__ X,
                                                 const float* __restrict__ w,
                                                 float* __restrict__ Y) {
  int row = blockIdx.x;
  int tid = threadIdx.x;
  const float* xr = X + (size_t)row * DM;
  float p = 0.f;
  for (int j = tid; j < DM; j += 256) { float v = xr[j]; p += v * v; }
  __shared__ float red[256];
  red[tid] = p;
  __syncthreads();
  for (int s = 128; s > 0; s >>= 1) {
    if (tid < s) red[tid] += red[tid + s];
    __syncthreads();
  }
  float scale = rsqrtf(red[0] / (float)DM + EPSF);
  float* yr = Y + (size_t)row * DM;
  for (int j = tid; j < DM; j += 256) yr[j] = xr[j] * scale * w[j];
}

// ---------------------------------------------------------------------------
// WMMA GEMM:  C[M,N] = act( A[M,K] * W[N,K]^T + bias ), optional C += ...
// A row stride lda, W row stride ldw (both K-contiguous => "NT" layout).
// Wave tile: 16(M) x 64(N) via 4 accumulators; block = 4 waves -> 32 x 128.
// N-edge handling: W row pointer clamped to N-1 BEFORE the K loop (uniform,
// branch-free inner loop); out-of-range columns are masked at the store.
// act: 0 = none, 1 = softplus. accum: 1 => C += result (residual add).
// ---------------------------------------------------------------------------
__global__ void __launch_bounds__(128) k_gemm(const float* __restrict__ A, int lda,
                                              const float* __restrict__ W, int ldw,
                                              float* __restrict__ C,
                                              int M, int N, int K,
                                              const float* __restrict__ bias,
                                              int act, int accum) {
  const int lane = threadIdx.x & 31;
  const int wave = threadIdx.x >> 5;
  const int wm = wave >> 1, wn = wave & 1;
  const int m0 = blockIdx.y * 32 + wm * 16;
  const int n0 = blockIdx.x * 128 + wn * 64;
  const int half = lane >> 4;    // which 16-lane half of the wave
  const int l16  = lane & 15;
  if (m0 >= M) return;

  v8f acc[4];
#pragma unroll
  for (int j = 0; j < 4; ++j)
    acc[j] = (v8f){0.f, 0.f, 0.f, 0.f, 0.f, 0.f, 0.f, 0.f};

  const float* Arow = A + (size_t)(m0 + l16) * lda;
  // Clamped B-operand row pointers (uniform inner loop; garbage columns are
  // never stored).
  const float* Wr[4];
#pragma unroll
  for (int j = 0; j < 4; ++j) {
    int wrow = n0 + j * 16 + l16;
    if (wrow > N - 1) wrow = N - 1;
    Wr[j] = W + (size_t)wrow * ldw;
  }

  const int kfull = K & ~31;
  int k0 = 0;
  for (; k0 < kfull; k0 += 32) {
    __builtin_prefetch(Arow + k0 + 128, 0, 0);       // global_prefetch_b8
    // A fragment (16x32 bf16): lanes 0-15 hold K = k0+{0..15},
    // lanes 16-31 hold K = k0+{16..31} for row m0+l16.
    v16bf a;
    const int ka = k0 + half * 16;
#pragma unroll
    for (int i = 0; i < 8; ++i) a[i]     = (__bf16)Arow[ka + i];
#pragma unroll
    for (int i = 0; i < 8; ++i) a[8 + i] = (__bf16)Arow[ka + 8 + i];

    // B fragments (32x16 bf16): lane holds column n = n0+16j+l16,
    // K = k0 + half*16 + {0..15} (contiguous along W's row).
    const int kb = k0 + half * 16;
#pragma unroll
    for (int j = 0; j < 4; ++j) {
      const float* p = Wr[j] + kb;
      v16bf bb;
#pragma unroll
      for (int i = 0; i < 16; ++i) bb[i] = (__bf16)p[i];
      acc[j] = __builtin_amdgcn_wmma_f32_16x16x32_bf16(
          false, a, false, bb, (short)0, acc[j], false, false);
    }
  }
  if (k0 < K) {  // K tail (e.g. K=48 for dt_proj), fully guarded on K
    v16bf a;
#pragma unroll
    for (int i = 0; i < 8; ++i) {
      int kk = k0 + half * 16 + i;
      a[i] = (__bf16)((kk < K) ? Arow[kk] : 0.f);
    }
#pragma unroll
    for (int i = 0; i < 8; ++i) {
      int kk = k0 + half * 16 + 8 + i;
      a[8 + i] = (__bf16)((kk < K) ? Arow[kk] : 0.f);
    }
#pragma unroll
    for (int j = 0; j < 4; ++j) {
      v16bf bb;
#pragma unroll
      for (int i = 0; i < 16; ++i) {
        int kk = k0 + half * 16 + i;
        bb[i] = (__bf16)((kk < K) ? Wr[j][kk] : 0.f);
      }
      acc[j] = __builtin_amdgcn_wmma_f32_16x16x32_bf16(
          false, a, false, bb, (short)0, acc[j], false, false);
    }
  }

  // Store: C/D layout => VGPR r: lanes 0-15 row m0+r, lanes 16-31 row m0+8+r.
#pragma unroll
  for (int j = 0; j < 4; ++j) {
    const int col = n0 + j * 16 + l16;
    if (col >= N) continue;
#pragma unroll
    for (int r = 0; r < 8; ++r) {
      const int row = m0 + half * 8 + r;
      float v = acc[j][r];
      if (bias) v += bias[col];
      if (act == 1) v = (v > 20.f) ? v : log1pf(__expf(v));  // softplus
      const size_t idx = (size_t)row * N + col;
      if (accum) C[idx] += v; else C[idx] = v;
    }
  }
}

// ---------------------------------------------------------------------------
// Depthwise causal conv (DCC=4) + bias + SiLU, reading xin = xz[:, :, :DI].
// ---------------------------------------------------------------------------
__global__ void __launch_bounds__(256) k_conv_silu(const float* __restrict__ xz,
                                                   const float* __restrict__ cw,
                                                   const float* __restrict__ cb,
                                                   float* __restrict__ xc) {
  int idx = blockIdx.x * 256 + threadIdx.x;            // over BB*LL*DI
  if (idx >= BB * LL * DI) return;
  int d = idx % DI;
  int t = idx / DI;                                    // b*LL + l
  int l = t % LL;
  float acc = cb[d];
#pragma unroll
  for (int q = 0; q < DCC; ++q) {
    int ll = l - (DCC - 1) + q;
    if (ll >= 0)
      acc += xz[(size_t)(t - l + ll) * (2 * DI) + d] * cw[d * DCC + q];
  }
  xc[(size_t)idx] = acc / (1.f + __expf(-acc));        // silu
}

// ---------------------------------------------------------------------------
// Selective scan + skip (u*D) + output gating with silu(z).
// One thread per (b,d) channel; 16-wide state in registers. The shared
// B/C vectors (32 floats per step) are broadcast into LDS with an async
// global->LDS copy (ASYNCcnt) issued by 8 lanes of wave 0.
// ---------------------------------------------------------------------------
__global__ void __launch_bounds__(256) k_scan(const float* __restrict__ delta,
                                              const float* __restrict__ dbc,
                                              const float* __restrict__ xc,
                                              const float* __restrict__ xz,
                                              const float* __restrict__ Alog,
                                              const float* __restrict__ Dw,
                                              float* __restrict__ yg) {
  const int tid = threadIdx.x;
  const int d = blockIdx.x * 256 + tid;
  const int b = blockIdx.y;

  float a[DSS];
#pragma unroll
  for (int s = 0; s < DSS; ++s) a[s] = -__expf(Alog[(size_t)d * DSS + s]);
  const float Dd = Dw[d];

  float h[DSS];
#pragma unroll
  for (int s = 0; s < DSS; ++s) h[s] = 0.f;

  __shared__ float sBC[2 * DSS];   // [0..15] = B, [16..31] = C
#if USE_ASYNC_LDS
  // LDS byte offset of sBC (low 32 bits of the LDS-aperture flat address).
  const unsigned lds_base = (unsigned)(uintptr_t)(void*)sBC;
#endif

  for (int l = 0; l < LL; ++l) {
    const int row = b * LL + l;
    __syncthreads();
#if USE_ASYNC_LDS
    if (tid < 8) {
      unsigned lds_addr = lds_base + (unsigned)tid * 16u;
      const void* gaddr =
          (const void*)((const char*)(dbc + (size_t)row * (DRR + 2 * DSS) + DRR) +
                        tid * 16);
      asm volatile("global_load_async_to_lds_b128 %0, %1, off"
                   :
                   : "v"(lds_addr), "v"(gaddr)
                   : "memory");
      asm volatile("s_wait_asynccnt 0x0" ::: "memory");
    }
#else
    if (tid < 2 * DSS)
      sBC[tid] = dbc[(size_t)row * (DRR + 2 * DSS) + DRR + tid];
#endif
    __syncthreads();

    const float dl = delta[(size_t)row * DI + d];
    const float u  = xc[(size_t)row * DI + d];
    const float z  = xz[(size_t)row * (2 * DI) + DI + d];
    const float du = dl * u;
    float y = 0.f;
#pragma unroll
    for (int s = 0; s < DSS; ++s) {
      h[s] = __expf(dl * a[s]) * h[s] + du * sBC[s];
      y += h[s] * sBC[DSS + s];
    }
    y += u * Dd;
    y *= z / (1.f + __expf(-z));                        // * silu(z)
    yg[(size_t)row * DI + d] = y;
  }
}

// ---------------------------------------------------------------------------
// Head: topic logits for pooled token 0, plus masked-mean NLL loss.
// ---------------------------------------------------------------------------
__global__ void __launch_bounds__(256) k_head(const float* __restrict__ xf,
                                              const float* __restrict__ topic_w,
                                              const float* __restrict__ topic_b,
                                              const int* __restrict__ labels,
                                              float* __restrict__ out) {
  __shared__ float logits[6];
  const int tid = threadIdx.x;
  const int pair = tid >> 5;
  const int lane = tid & 31;
  if (pair < 6) {
    const int b = pair / 3, c = pair % 3;
    const float* row = xf + (size_t)b * LL * DM;        // token l = 0
    const float* w = topic_w + (size_t)c * DM;
    float s = 0.f;
    for (int j = lane; j < DM; j += 32) s += row[j] * w[j];
    for (int off = 16; off > 0; off >>= 1) s += __shfl_xor(s, off, 32);
    if (lane == 0) {
      float v = s + topic_b[c];
      logits[pair] = v;
      out[b * 3 + c] = v;
    }
  }
  __syncthreads();
  if (tid == 0) {
    float total = 0.f, cnt = 0.f;
    for (int b = 0; b < BB; ++b) {
      int lab = labels[b];
      float l0 = logits[b * 3 + 0], l1 = logits[b * 3 + 1], l2 = logits[b * 3 + 2];
      float m = fmaxf(l0, fmaxf(l1, l2));
      float lse = m + logf(__expf(l0 - m) + __expf(l1 - m) + __expf(l2 - m));
      float nll = lse - logits[b * 3 + lab];
      if (lab != 0) { total += nll; cnt += 1.f; }
    }
    out[6] = total / fmaxf(cnt, 1.f);
  }
}

// ---------------------------------------------------------------------------
// Host-side orchestration.
// ---------------------------------------------------------------------------
static inline void launch_gemm(const float* A, int lda, const float* W, int ldw,
                               float* C, int M, int N, int K,
                               const float* bias, int act, int accum,
                               hipStream_t s) {
  dim3 grid((N + 127) / 128, (M + 31) / 32);
  k_gemm<<<grid, 128, 0, s>>>(A, lda, W, ldw, C, M, N, K, bias, act, accum);
}

extern "C" void kernel_launch(void* const* d_in, const int* in_sizes, int n_in,
                              void* d_out, int out_size, void* d_ws, size_t ws_size,
                              hipStream_t stream) {
  (void)in_sizes; (void)n_in; (void)out_size; (void)ws_size;

  const int*   tokens    = (const int*)d_in[0];
  const int*   labels    = (const int*)d_in[1];
  const float* emb       = (const float*)d_in[2];
  const float* in_proj_w = (const float*)d_in[3];
  const float* conv_w    = (const float*)d_in[4];
  const float* conv_b    = (const float*)d_in[5];
  const float* x_proj_w  = (const float*)d_in[6];
  const float* dt_proj_w = (const float*)d_in[7];
  const float* dt_proj_b = (const float*)d_in[8];
  const float* A_log     = (const float*)d_in[9];
  const float* Dw        = (const float*)d_in[10];
  const float* out_proj_w= (const float*)d_in[11];
  const float* norm_w    = (const float*)d_in[12];
  const float* norm_f_w  = (const float*)d_in[13];
  const float* topic_w   = (const float*)d_in[14];
  const float* topic_b   = (const float*)d_in[15];
  // d_in[16], d_in[17] = sent_w, sent_b (unused: sentiment not returned)

  float* out = (float*)d_out;
  float* ws  = (float*)d_ws;

  const int T = BB * LL;                // 4096 token rows
  size_t o = 0;
  float* x     = ws + o; o += (size_t)T * DM;           // residual stream
  float* xn    = ws + o; o += (size_t)T * DM;           // rmsnorm(x)
  float* xz    = ws + o; o += (size_t)T * 2 * DI;       // in_proj out
  float* xc    = ws + o; o += (size_t)T * DI;           // conv+silu
  float* dbc   = ws + o; o += (size_t)T * (DRR + 2*DSS);// x_proj out
  float* delta = ws + o; o += (size_t)T * DI;           // softplus(dt_proj)
  float* yg    = ws + o; o += (size_t)T * DI;           // gated scan output

  // 1) Embedding gather
  {
    int n = T * NCC * DMC;
    k_embed<<<(n + 255) / 256, 256, 0, stream>>>(tokens, emb, x);
  }

  // 2) Mamba layers
  for (int layer = 0; layer < NLL; ++layer) {
    const float* W_in  = in_proj_w  + (size_t)layer * (2 * DI) * DM;
    const float* W_cv  = conv_w     + (size_t)layer * DI * DCC;
    const float* b_cv  = conv_b     + (size_t)layer * DI;
    const float* W_xp  = x_proj_w   + (size_t)layer * (DRR + 2 * DSS) * DI;
    const float* W_dt  = dt_proj_w  + (size_t)layer * DI * DRR;
    const float* b_dt  = dt_proj_b  + (size_t)layer * DI;
    const float* Al    = A_log      + (size_t)layer * DI * DSS;
    const float* Dl    = Dw         + (size_t)layer * DI;
    const float* W_out = out_proj_w + (size_t)layer * DM * DI;
    const float* w_nrm = norm_w     + (size_t)layer * DM;

    k_rmsnorm<<<T, 256, 0, stream>>>(x, w_nrm, xn);

    // xz = xn @ in_proj_w^T   (4096 x 3072, K=768)
    launch_gemm(xn, DM, W_in, DM, xz, T, 2 * DI, DM, nullptr, 0, 0, stream);

    // depthwise conv + SiLU on xin half
    {
      int n = T * DI;
      k_conv_silu<<<(n + 255) / 256, 256, 0, stream>>>(xz, W_cv, b_cv, xc);
    }

    // dbc = xc @ x_proj_w^T   (4096 x 80, K=1536)
    launch_gemm(xc, DI, W_xp, DI, dbc, T, DRR + 2 * DSS, DI, nullptr, 0, 0, stream);

    // delta = softplus(dt @ dt_proj_w^T + b)   (A = first 48 cols of dbc, lda=80)
    launch_gemm(dbc, DRR + 2 * DSS, W_dt, DRR, delta, T, DI, DRR, b_dt, 1, 0, stream);

    // selective scan + D skip + silu(z) gating
    {
      dim3 grid(DI / 256, BB);
      k_scan<<<grid, 256, 0, stream>>>(delta, dbc, xc, xz, Al, Dl, yg);
    }

    // x += yg @ out_proj_w^T   (residual, 4096 x 768, K=1536)
    launch_gemm(yg, DI, W_out, DI, x, T, DM, DI, nullptr, 0, 1, stream);
  }

  // 3) Final RMSNorm straight into the x-output slot of d_out
  float* x_out = out + 7;
  k_rmsnorm<<<T, 256, 0, stream>>>(x, norm_f_w, x_out);

  // 4) Head: topic logits + masked NLL loss
  k_head<<<1, 256, 0, stream>>>(x_out, topic_w, topic_b, labels, out);
}
